// BiChannelAttention_31258771980811
// MI455X (gfx1250) — compile-verified, matrix operand order checked by reading, not run on hardware
//
#include <hip/hip_runtime.h>
#include <math.h>

// ---------------------------------------------------------------------------
// BiChannelAttention decode step, MI455X (gfx1250, wave32, WMMA + TDM).
//
// Algebraic plan (exact w.r.t. the reference up to fp reassociation):
//   * time_mask adds a uniform -1e6 to every score -> cancelled by softmax.
//   * bk.q score bias is uniform over t -> cancelled by softmax.
//   * local window: out-of-window scores sit 1e6 below in-window ones;
//     expf underflows to exactly 0.0f -> only last 512 positions computed.
//   * fold K: score[t] = c[t] . (Wk^T q) / sqrt(96)
//   * fold V: out = (sum_t attn[t] * c[t]) @ Wv + bv     (sum attn == 1)
// Kernel 1: per-head WMMA, Q = U@Wq+b (16x64x64), W = Q@Wk^T  -> ws
// Kernel 2: per-(b,h): TENSOR_LOAD_TO_LDS stages the 511x96 window tile,
//           then scores, softmax, attn-weighted row sum — one HBM pass.
// Kernel 3: per-head WMMA, Out = S@Wv + bv + content_t residual.
// ---------------------------------------------------------------------------

#define H_   16
#define DU_  64
#define DP_  32
#define DD_  96
#define FF_  (H_*DD_)      // 1536
#define BB_  16
#define TC_  4095
#define TT_  (TC_+1)       // 4096
#define WIN_ 512
#define T0_  (TT_-WIN_)    // 3584
#define INV_SQRTD 0.10206207261596577f   // 1/sqrt(96)

#if __has_builtin(__builtin_amdgcn_tensor_load_to_lds) && \
    __has_builtin(__builtin_amdgcn_s_wait_tensorcnt)
#define HAVE_TDM_ 1
#else
#define HAVE_TDM_ 0
#endif

typedef __attribute__((ext_vector_type(16))) __bf16 v16bf;
typedef __attribute__((ext_vector_type(8)))  float  v8f;
typedef __attribute__((ext_vector_type(4)))  unsigned int v4u;
typedef __attribute__((ext_vector_type(8)))  int    v8i;
typedef __attribute__((ext_vector_type(4)))  int    v4i;

__device__ __forceinline__ v8f wmma_bf16(v16bf a, v16bf b, v8f c) {
  // D = A(16x32,bf16) x B(32x16,bf16) + C(16x16,f32)
  return __builtin_amdgcn_wmma_f32_16x16x32_bf16(
      /*neg_a=*/false, a, /*neg_b=*/false, b,
      /*c_mod=*/(short)0, c, /*reuse_a=*/false, /*reuse_b=*/false);
}

// A fragment: source is row-major 16xK (ld = row stride), K-slice [kk,kk+32).
// wave32 layout: lane 0-15 -> M=lane, K in {kk..kk+7, kk+16..kk+23};
//                lane 16-31 -> M=lane-16, K in {kk+8..kk+15, kk+24..kk+31}.
__device__ __forceinline__ v16bf afrag(const float* src, int ld, int kk, int lane) {
  const int m  = lane & 15;
  const int kb = (lane >> 4) ? 8 : 0;
  v16bf a;
#pragma unroll
  for (int i = 0; i < 16; ++i) {
    const int k = kk + kb + ((i < 8) ? i : (8 + i));
    a[i] = (__bf16)src[m * ld + k];
  }
  return a;
}

// B fragment from row-major KxN source: element (k, n=nt*16+lane%16).
__device__ __forceinline__ v16bf bfrag(const float* src, int ld, int kk, int nt, int lane) {
  const int n  = (lane & 15) + nt * 16;
  const int kb = (lane >> 4) ? 8 : 0;
  v16bf b;
#pragma unroll
  for (int i = 0; i < 16; ++i) {
    const int k = kk + kb + ((i < 8) ? i : (8 + i));
    b[i] = (__bf16)src[k * ld + n];
  }
  return b;
}

// B fragment of W^T where W is row-major NxK: element (k,n) = W[n*ld + k].
__device__ __forceinline__ v16bf bfragT(const float* src, int ld, int kk, int nt, int lane) {
  const int n  = (lane & 15) + nt * 16;
  const int kb = (lane >> 4) ? 8 : 0;
  v16bf b;
#pragma unroll
  for (int i = 0; i < 16; ++i) {
    const int k = kk + kb + ((i < 8) ? i : (8 + i));
    b[i] = (__bf16)src[n * ld + k];
  }
  return b;
}

// ---------------------------------------------------------------------------
// Kernel 1: per head h, one wave. Q_u/Q_p then folded weights W = Q @ Wk^T.
// wvec layout: (b, h, 96) with u-part in [0,64), p-part in [64,96).
// ---------------------------------------------------------------------------
__global__ __launch_bounds__(32) void k_fold_qw(
    const float* __restrict__ content,
    const float* __restrict__ Wq_u, const float* __restrict__ bq_u,
    const float* __restrict__ Wk_u,
    const float* __restrict__ Wq_p, const float* __restrict__ bq_p,
    const float* __restrict__ Wk_p,
    float* __restrict__ wvec)
{
  const int h    = blockIdx.x;
  const int lane = threadIdx.x;

  __shared__ float sU[16 * 64], sPm[16 * 32];
  __shared__ float sWqu[64 * 64], sWku[64 * 64];
  __shared__ float sWqp[32 * 32], sWkp[32 * 32];
  __shared__ float sbu[64], sbp[32];
  __shared__ float sQu[16 * 64], sQp[16 * 32];

  for (int i = lane; i < 16 * 64; i += 32) {
    const int m = i >> 6, d = i & 63;
    sU[i] = content[m * FF_ + h * DD_ + d];
  }
  for (int i = lane; i < 16 * 32; i += 32) {
    const int m = i >> 5, d = i & 31;
    sPm[i] = content[m * FF_ + h * DD_ + DU_ + d];
  }
  for (int i = lane; i < 64 * 64; i += 32) {
    sWqu[i] = Wq_u[h * 4096 + i];
    sWku[i] = Wk_u[h * 4096 + i];
  }
  for (int i = lane; i < 32 * 32; i += 32) {
    sWqp[i] = Wq_p[h * 1024 + i];
    sWkp[i] = Wk_p[h * 1024 + i];
  }
  for (int i = lane; i < 64; i += 32) sbu[i] = bq_u[h * 64 + i];
  if (lane < 32) sbp[lane] = bq_p[h * 32 + lane];
  __syncthreads();

  const int nD    = lane & 15;           // C/D: lane -> column
  const int mBase = (lane >> 4) ? 8 : 0; // C/D: vgpr r -> row mBase+r

  // Q_u = U @ Wq_u + bq_u   (16x64, K=64)
#pragma unroll
  for (int nt = 0; nt < 4; ++nt) {
    v8f acc = {};
#pragma unroll
    for (int kk = 0; kk < 64; kk += 32)
      acc = wmma_bf16(afrag(sU, 64, kk, lane), bfrag(sWqu, 64, kk, nt, lane), acc);
    const int col = nt * 16 + nD;
#pragma unroll
    for (int r = 0; r < 8; ++r)
      sQu[(mBase + r) * 64 + col] = acc[r] + sbu[col];
  }
  // Q_p = P @ Wq_p + bq_p   (16x32, K=32)
#pragma unroll
  for (int nt = 0; nt < 2; ++nt) {
    v8f z = {};
    v8f acc = wmma_bf16(afrag(sPm, 32, 0, lane), bfrag(sWqp, 32, 0, nt, lane), z);
    const int col = nt * 16 + nD;
#pragma unroll
    for (int r = 0; r < 8; ++r)
      sQp[(mBase + r) * 32 + col] = acc[r] + sbp[col];
  }
  __syncthreads();

  // w_u = Q_u @ Wk_u^T, pre-scaled by 1/sqrt(96)
#pragma unroll
  for (int nt = 0; nt < 4; ++nt) {
    v8f acc = {};
#pragma unroll
    for (int kk = 0; kk < 64; kk += 32)
      acc = wmma_bf16(afrag(sQu, 64, kk, lane), bfragT(sWku, 64, kk, nt, lane), acc);
    const int col = nt * 16 + nD;
#pragma unroll
    for (int r = 0; r < 8; ++r)
      wvec[((mBase + r) * H_ + h) * DD_ + col] = acc[r] * INV_SQRTD;
  }
  // w_p = Q_p @ Wk_p^T
#pragma unroll
  for (int nt = 0; nt < 2; ++nt) {
    v8f z = {};
    v8f acc = wmma_bf16(afrag(sQp, 32, 0, lane), bfragT(sWkp, 32, 0, nt, lane), z);
    const int col = nt * 16 + nD;
#pragma unroll
    for (int r = 0; r < 8; ++r)
      wvec[((mBase + r) * H_ + h) * DD_ + DU_ + col] = acc[r] * INV_SQRTD;
  }
}

// ---------------------------------------------------------------------------
// Kernel 2: per (b,h) block (256 thr = 8 waves).
//   Stage the 511x96 f32 cache window into LDS with one TDM descriptor
//   (tile_dim0=96, tile_dim1=511, tensor_dim0_stride=1536, data_size=4B),
//   append the current token row, then: scores -> softmax -> weighted sum.
//   One HBM traversal; Phase C reads the LDS tile.
// ---------------------------------------------------------------------------
__global__ __launch_bounds__(256) void k_attend(
    const float* __restrict__ content,
    const float* __restrict__ cache,
    const float* __restrict__ wvec,
    const float* __restrict__ pos_param,
    float* __restrict__ svec)
{
  const int h    = blockIdx.x;
  const int b    = blockIdx.y;
  const int tid  = threadIdx.x;
  const int wv   = tid >> 5;
  const int lane = tid & 31;

  __shared__ __align__(16) float sTile[WIN_ * DD_];   // 512 x 96 f32 (196.6 KB)
  __shared__ float sSc[WIN_];
  __shared__ float sRed[256];
  __shared__ float sPart[8 * 96];

  const float* base = cache + ((size_t)b * TC_ + T0_) * FF_ + (size_t)h * DD_;
  const float* cur  = content + (size_t)b * FF_ + h * DD_;

#if HAVE_TDM_
  // ---- TDM: one 2D tile load, issued by wave 0 only (scalar branch) ----
  if (__builtin_amdgcn_readfirstlane(tid) == 0) {
    const unsigned long long ga = (unsigned long long)(uintptr_t)base;
    const unsigned lds_off = (unsigned)(uintptr_t)&sTile[0];  // low 32b = LDS byte addr
    v4u g0;
    g0.x = 1u;                                   // count=1, user mode, no gather
    g0.y = lds_off;                              // D#.lds_addr  (bits 63:32)
    g0.z = (unsigned)(ga & 0xFFFFFFFFu);         // D#.global_addr[31:0]
    g0.w = (unsigned)((ga >> 32) & 0x01FFFFFFu)  // D#.global_addr[56:32]
         | (2u << 30);                           // D#.type = 2 (image)
    v8i g1;
    g1[0] = (int)(2u << 16);        // workgroup_mask=0, data_size=4B
    g1[1] = (int)(96u << 16);       // tensor_dim0[15:0]=96 (atomic_barrier=0)
    g1[2] = (int)(511u << 16);      // tensor_dim0[31:16]=0, tensor_dim1[15:0]=511
    g1[3] = (int)(96u << 16);       // tensor_dim1[31:16]=0, tile_dim0=96
    g1[4] = (int)511;               // tile_dim1=511, tile_dim2=0 (2D)
    g1[5] = (int)1536;              // tensor_dim0_stride[31:0] = F
    g1[6] = 0;                      // stride0[47:32]=0, stride1[15:0]=0
    g1[7] = 0;                      // tensor_dim1_stride hi = 0
    v4i z4 = {0, 0, 0, 0};
#if __has_include(<hip/amd_detail/amd_gfx1250_TDM.h>)
    v8i z8 = {0, 0, 0, 0, 0, 0, 0, 0};
    __builtin_amdgcn_tensor_load_to_lds(g0, g1, z4, z4, z8, 0);   // 6-arg toolchain
#else
    __builtin_amdgcn_tensor_load_to_lds(g0, g1, z4, z4, 0);       // ROCm 7.2 (5-arg)
#endif
    __builtin_amdgcn_s_wait_tensorcnt(0);
  }
#else
  // ---- Fallback: cooperative coalesced copy of the 511 cache rows ----
  for (int i = tid; i < (WIN_ - 1) * DD_; i += 256) {
    const int r = i / DD_, c = i - r * DD_;
    sTile[i] = base[(size_t)r * FF_ + c];
  }
#endif
  // Row 511 = current token (content_t head slice), both modes.
  if (tid < DD_) sTile[(WIN_ - 1) * DD_ + tid] = cur[tid];
  __syncthreads();

  const int   wb = (b * H_ + h) * DD_;
  const float w0 = wvec[wb + lane];
  const float w1 = wvec[wb + lane + 32];
  const float w2 = wvec[wb + lane + 64];
  const float pp = pos_param[0];
  const float pos_scale = 16.0f / logf(8.0f);

  // ---- Phase A: scores (lane l covers dims l, l+32, l+64 of the 96) ----
  for (int i = 0; i < 64; ++i) {
    const int r = wv * 64 + i;
    const float* row = &sTile[r * DD_];
    float part = row[lane] * w0 + row[lane + 32] * w1 + row[lane + 64] * w2;
#pragma unroll
    for (int off = 16; off; off >>= 1) part += __shfl_xor(part, off, 32);
    if (lane == 0) {
      const int n = (WIN_ - 1) - r;             // T5 relative distance
      float pb;
      if (n < 16) {
        pb = (float)n;
      } else {
        int lb = 16 + (int)(logf((float)n * (1.0f / 16.0f)) * pos_scale);
        pb = (float)(lb > 31 ? 31 : lb);
      }
      sSc[r] = part - pp * pb;                  // uniform mask terms dropped
    }
  }
  __syncthreads();

  // ---- Phase B: softmax over 512 ----
  float v = fmaxf(sSc[tid], sSc[tid + 256]);
  sRed[tid] = v;
  __syncthreads();
#pragma unroll
  for (int s = 128; s > 0; s >>= 1) {
    if (tid < s) sRed[tid] = fmaxf(sRed[tid], sRed[tid + s]);
    __syncthreads();
  }
  const float mx = sRed[0];
  __syncthreads();
  const float e0 = __expf(sSc[tid] - mx);
  const float e1 = __expf(sSc[tid + 256] - mx);
  sSc[tid] = e0;
  sSc[tid + 256] = e1;
  sRed[tid] = e0 + e1;
  __syncthreads();
#pragma unroll
  for (int s = 128; s > 0; s >>= 1) {
    if (tid < s) sRed[tid] += sRed[tid + s];
    __syncthreads();
  }
  const float denom = sRed[0];
  __syncthreads();

  // ---- Phase C: s[d] = sum_r attn[r]*c[r,d], straight from the LDS tile ----
  float a0 = 0.f, a1 = 0.f, a2 = 0.f;
  for (int i = 0; i < 64; ++i) {
    const int r = wv * 64 + i;
    const float* row = &sTile[r * DD_];
    const float e = sSc[r];
    a0 += e * row[lane];
    a1 += e * row[lane + 32];
    a2 += e * row[lane + 64];
  }
  sPart[wv * 96 + lane]      = a0;
  sPart[wv * 96 + lane + 32] = a1;
  sPart[wv * 96 + lane + 64] = a2;
  __syncthreads();
  if (tid < 96) {
    float s = 0.f;
#pragma unroll
    for (int w = 0; w < 8; ++w) s += sPart[w * 96 + tid];
    svec[(b * H_ + h) * DD_ + tid] = s / denom;
  }
}

// ---------------------------------------------------------------------------
// Kernel 3: per head h, one wave. Out = S @ Wv + bv + content_t residual.
// ---------------------------------------------------------------------------
__global__ __launch_bounds__(32) void k_project_out(
    const float* __restrict__ content,
    const float* __restrict__ svec,
    const float* __restrict__ Wv_u, const float* __restrict__ bv_u,
    const float* __restrict__ Wv_p, const float* __restrict__ bv_p,
    float* __restrict__ out)
{
  const int h    = blockIdx.x;
  const int lane = threadIdx.x;

  __shared__ float sS[16 * 96];
  __shared__ float sWu[64 * 64], sWp[32 * 32];
  __shared__ float sbu[64], sbp[32];

  for (int i = lane; i < 16 * 96; i += 32) {
    const int m = i / 96, d = i % 96;
    sS[i] = svec[(m * H_ + h) * DD_ + d];
  }
  for (int i = lane; i < 64 * 64; i += 32) sWu[i] = Wv_u[h * 4096 + i];
  for (int i = lane; i < 32 * 32; i += 32) sWp[i] = Wv_p[h * 1024 + i];
  for (int i = lane; i < 64; i += 32) sbu[i] = bv_u[h * 64 + i];
  if (lane < 32) sbp[lane] = bv_p[h * 32 + lane];
  __syncthreads();

  const int nD    = lane & 15;
  const int mBase = (lane >> 4) ? 8 : 0;

  // u-channel: 16x64, K=64
#pragma unroll
  for (int nt = 0; nt < 4; ++nt) {
    v8f acc = {};
#pragma unroll
    for (int kk = 0; kk < 64; kk += 32)
      acc = wmma_bf16(afrag(sS, 96, kk, lane), bfrag(sWu, 64, kk, nt, lane), acc);
    const int col = nt * 16 + nD;
#pragma unroll
    for (int r = 0; r < 8; ++r) {
      const int m = mBase + r;
      out[m * FF_ + h * DD_ + col] =
          acc[r] + sbu[col] + content[m * FF_ + h * DD_ + col];
    }
  }
  // p-channel: 16x32, K=32 (A = sS columns 64..95)
#pragma unroll
  for (int nt = 0; nt < 2; ++nt) {
    v8f z = {};
    v8f acc = wmma_bf16(afrag(sS + DU_, 96, 0, lane), bfrag(sWp, 32, 0, nt, lane), z);
    const int col = nt * 16 + nD;
#pragma unroll
    for (int r = 0; r < 8; ++r) {
      const int m = mBase + r;
      out[m * FF_ + h * DD_ + DU_ + col] =
          acc[r] + sbp[col] + content[m * FF_ + h * DD_ + DU_ + col];
    }
  }
}

// ---------------------------------------------------------------------------
extern "C" void kernel_launch(void* const* d_in, const int* in_sizes, int n_in,
                              void* d_out, int out_size, void* d_ws, size_t ws_size,
                              hipStream_t stream) {
  (void)in_sizes; (void)n_in; (void)out_size; (void)ws_size;

  // setup_inputs() order:
  // 0:t 1:content_t 2:time_mask 3:cache 4:speakers 5:Wq_u 6:bq_u 7:Wk_u 8:bk_u
  // 9:Wv_u 10:bv_u 11:Wq_p 12:bq_p 13:Wk_p 14:bk_p 15:Wv_p 16:bv_p 17:pos_param
  const float* content  = (const float*)d_in[1];
  const float* cache    = (const float*)d_in[3];
  const float* Wq_u     = (const float*)d_in[5];
  const float* bq_u     = (const float*)d_in[6];
  const float* Wk_u     = (const float*)d_in[7];
  // bk_u/bk_p: uniform score shift -> softmax-invariant, unused.
  // time_mask: reference double-where makes it a uniform -1e6 shift, unused.
  // speakers: never used by the reference.
  const float* Wv_u     = (const float*)d_in[9];
  const float* bv_u     = (const float*)d_in[10];
  const float* Wq_p     = (const float*)d_in[11];
  const float* bq_p     = (const float*)d_in[12];
  const float* Wk_p     = (const float*)d_in[13];
  const float* Wv_p     = (const float*)d_in[15];
  const float* bv_p     = (const float*)d_in[16];
  const float* pos_par  = (const float*)d_in[17];

  float* wvec = (float*)d_ws;                 // B*H*96 floats
  float* svec = wvec + BB_ * H_ * DD_;        // B*H*96 floats
  float* outp = (float*)d_out;                // B*F floats

  k_fold_qw<<<dim3(H_), dim3(32), 0, stream>>>(
      content, Wq_u, bq_u, Wk_u, Wq_p, bq_p, Wk_p, wvec);
  k_attend<<<dim3(H_, BB_), dim3(256), 0, stream>>>(
      content, cache, wvec, pos_par, svec);
  k_project_out<<<dim3(H_), dim3(32), 0, stream>>>(
      content, svec, Wv_u, bv_u, Wv_p, bv_p, outp);
}
